// Attention_23124103921869
// MI455X (gfx1250) — matrix operand order repaired and checked by reading, not compiled, not run on hardware
//
#include <hip/hip_runtime.h>
#include <hip/hip_bf16.h>
#include <stdint.h>

// Problem constants (B,S,D,H from the reference)
#define B_   4
#define S_   2048
#define D_   1024
#define H_   16
#define HD_  64
#define BS_  (B_ * S_)          // 8192 rows
#define N3D  (3 * D_)           // 3072
#define NEGINF (-1e9f)

typedef __attribute__((ext_vector_type(16))) __bf16 v16bf;
typedef __attribute__((ext_vector_type(8)))  float  v8f;

union Frag16 { v16bf v; uint4 q[2]; __bf16 e[16]; };
union Acc8   { v8f   v; float f[8]; };

static __device__ __forceinline__ v8f wmma_bf16(v16bf a, v16bf b, v8f c) {
  // D = A(16x32 bf16) * B(32x16 bf16) + C(16x16 f32)
  return __builtin_amdgcn_wmma_f32_16x16x32_bf16(false, a, false, b,
                                                 (short)0, c, false, false);
}

// ----- CDNA5 async global->LDS copy (guarded; falls back to sync path) -----
#if __has_builtin(__builtin_amdgcn_global_load_async_to_lds_b128)
#define HAVE_ASYNC_LDS 1
// Parameter types per hipcc diagnostic: int __vector_size__(16) pointers,
// first in the global (AS1 / __device__) space, second in LDS (AS3).
typedef int b128_t __attribute__((vector_size(16)));
typedef __attribute__((address_space(1))) b128_t as1_b128;
typedef __attribute__((address_space(3))) b128_t as3_b128;
static __device__ __forceinline__ void async_copy16(const void* g, void* l) {
  __builtin_amdgcn_global_load_async_to_lds_b128((as1_b128*)g, (as3_b128*)l, 0, 0);
}
static __device__ __forceinline__ void wait_async() {
#if __has_builtin(__builtin_amdgcn_s_wait_asynccnt)
  __builtin_amdgcn_s_wait_asynccnt(0);
#else
  asm volatile("s_wait_asynccnt 0" ::: "memory");
#endif
}
#else
#define HAVE_ASYNC_LDS 0
#endif

// Build a bf16 A-fragment from an fp32 LDS row (convert at fragment build)
static __device__ __forceinline__ v16bf make_afrag_f32(const float* base, int g) {
  const float4 c0 = *(const float4*)(base + 8 * g);
  const float4 c1 = *(const float4*)(base + 8 * g + 4);
  const float4 c2 = *(const float4*)(base + 16 + 8 * g);
  const float4 c3 = *(const float4*)(base + 16 + 8 * g + 4);
  Frag16 f;
  f.e[0]  = (__bf16)c0.x; f.e[1]  = (__bf16)c0.y;
  f.e[2]  = (__bf16)c0.z; f.e[3]  = (__bf16)c0.w;
  f.e[4]  = (__bf16)c1.x; f.e[5]  = (__bf16)c1.y;
  f.e[6]  = (__bf16)c1.z; f.e[7]  = (__bf16)c1.w;
  f.e[8]  = (__bf16)c2.x; f.e[9]  = (__bf16)c2.y;
  f.e[10] = (__bf16)c2.z; f.e[11] = (__bf16)c2.w;
  f.e[12] = (__bf16)c3.x; f.e[13] = (__bf16)c3.y;
  f.e[14] = (__bf16)c3.z; f.e[15] = (__bf16)c3.w;
  return f.v;
}

// ---------------------------------------------------------------------------
// Kernel 1: QKV projection  x[8192,1024] @ Wqkv[1024,3072] + bqkv
// 128x128 tile per 256-thread block, BK=64 (16 WMMAs per barrier round).
// A tile staged fp32 via async global->LDS; converted to bf16 at frag build.
// Writes Q (pre-scaled by HD^-0.5), K, V as bf16 in [B,H,S,HD] layout.
// ---------------------------------------------------------------------------
__global__ __launch_bounds__(256) void qkv_gemm_kernel(
    const float* __restrict__ x, const float* __restrict__ Wqkv,
    const float* __restrict__ bqkv,
    unsigned short* __restrict__ qws_u, unsigned short* __restrict__ kws_u,
    unsigned short* __restrict__ vws_u) {
  __shared__ __align__(16) float  As[128][72];   // 128 x 64 fp32 (pitch 288B)
  __shared__ __align__(16) __bf16 Bt[128][80];   // B transposed [n][k] (pitch 160B)

  const int tid  = threadIdx.x;
  const int lane = tid & 31, wv = tid >> 5;
  const int g = lane >> 4, ln = lane & 15;
  const int wr = wv >> 1, wc = wv & 1;           // wave grid 4x2
  const int n0 = blockIdx.x * 128;
  const int m0 = blockIdx.y * 128;

  Acc8 acc[2][4];
#pragma unroll
  for (int i = 0; i < 2; ++i)
#pragma unroll
    for (int j = 0; j < 4; ++j)
#pragma unroll
      for (int r = 0; r < 8; ++r) acc[i][j].f[r] = 0.f;

  for (int k0 = 0; k0 < D_; k0 += 64) {
    // --- A tile: 128x64 fp32, async DMA into LDS ---
    {
      const int r0 = tid >> 4, c4 = tid & 15;
#pragma unroll
      for (int rr = 0; rr < 8; ++rr) {
        const int row = rr * 16 + r0;
        const float* src = x + (size_t)(m0 + row) * D_ + k0 + c4 * 4;
#if HAVE_ASYNC_LDS
        async_copy16(src, &As[row][c4 * 4]);
#else
        *(float4*)&As[row][c4 * 4] = *(const float4*)src;
#endif
      }
    }
    // --- B tile: 64x128 fp32 -> bf16, transposed into LDS ---
    {
      const int r0 = tid >> 5, c4 = tid & 31;
#pragma unroll
      for (int rr = 0; rr < 8; ++rr) {
        const int row = rr * 8 + r0;
        const float4 f = *(const float4*)(Wqkv + (size_t)(k0 + row) * N3D + n0 + c4 * 4);
        Bt[c4 * 4 + 0][row] = (__bf16)f.x;
        Bt[c4 * 4 + 1][row] = (__bf16)f.y;
        Bt[c4 * 4 + 2][row] = (__bf16)f.z;
        Bt[c4 * 4 + 3][row] = (__bf16)f.w;
      }
    }
#if HAVE_ASYNC_LDS
    wait_async();
#endif
    __syncthreads();

#pragma unroll
    for (int kk = 0; kk < 64; kk += 32) {
      Frag16 fa[2], fb[4];
#pragma unroll
      for (int ti = 0; ti < 2; ++ti)
        fa[ti].v = make_afrag_f32(&As[wr * 32 + ti * 16 + ln][kk], g);
#pragma unroll
      for (int tj = 0; tj < 4; ++tj) {
        const __bf16* base = &Bt[wc * 64 + tj * 16 + ln][kk];
        fb[tj].q[0] = *(const uint4*)(base + 8 * g);      // K = kk+8g..+7, kk+16+8g..+7
        fb[tj].q[1] = *(const uint4*)(base + 16 + 8 * g);
      }
#pragma unroll
      for (int ti = 0; ti < 2; ++ti)
#pragma unroll
        for (int tj = 0; tj < 4; ++tj)
          acc[ti][tj].v = wmma_bf16(fa[ti].v, fb[tj].v, acc[ti][tj].v);
    }
    __syncthreads();
  }

  // --- epilogue: +bias, scale Q, scatter to [B,H,S,HD] bf16 ---
#pragma unroll
  for (int tj = 0; tj < 4; ++tj) {
    const int nc  = n0 + wc * 64 + tj * 16 + ln;
    const int sel = nc >> 10;                 // 0=Q 1=K 2=V
    const int dd  = nc & 1023;
    const int h   = dd >> 6, hd = dd & 63;
    const float bias  = bqkv[nc];
    const float scale = (sel == 0) ? 0.125f : 1.0f;  // HD^-0.5 folded into Q
    __bf16* dst = (__bf16*)(sel == 0 ? qws_u : (sel == 1 ? kws_u : vws_u));
#pragma unroll
    for (int ti = 0; ti < 2; ++ti)
#pragma unroll
      for (int r = 0; r < 8; ++r) {
        const int mr = m0 + wr * 32 + ti * 16 + r + 8 * g;
        const int bb = mr >> 11, s = mr & 2047;
        const float v = (acc[ti][tj].f[r] + bias) * scale;
        dst[((size_t)((bb * H_ + h) * S_ + s)) * HD_ + hd] = (__bf16)v;
      }
  }
}

// ---------------------------------------------------------------------------
// Kernel 2: flash attention.  Grid: x = b*h (64), y = 128-query tile (16).
// 8 waves x 16 query rows each; 64-key blocks; online softmax in fp32.
// rel_bias streams straight into the WMMA C operand (dominant HBM term).
// ---------------------------------------------------------------------------
__global__ __launch_bounds__(256) void flash_attn_kernel(
    const unsigned short* __restrict__ qws_u, const unsigned short* __restrict__ kws_u,
    const unsigned short* __restrict__ vws_u, const float* __restrict__ rel_bias,
    const unsigned char* __restrict__ pad, unsigned short* __restrict__ ows_u) {
  __shared__ __align__(16) __bf16 Vt[64][80];       // V transposed: [hd][key]
  __shared__ __align__(16) __bf16 Pl[8][16][80];    // per-wave P tile

  const int tid  = threadIdx.x;
  const int lane = tid & 31, wv = tid >> 5;
  const int g = lane >> 4, ln = lane & 15;
  const int bh = blockIdx.x, b = bh >> 4, h = bh & 15;
  const int q0 = blockIdx.y * 128;
  const int qbase = q0 + wv * 16;

  const __bf16* Qp = (const __bf16*)qws_u + ((size_t)(b * H_ + h)) * S_ * HD_;
  const __bf16* Kp = (const __bf16*)kws_u + ((size_t)(b * H_ + h)) * S_ * HD_;
  const __bf16* Vp = (const __bf16*)vws_u + ((size_t)(b * H_ + h)) * S_ * HD_;
  const float*  relp = rel_bias + (size_t)h * S_ * S_;

  // Q A-fragments (2 k-steps over HD=64), loaded once from global
  Frag16 qa[2];
  {
    const __bf16* qrow = Qp + (size_t)(qbase + ln) * HD_;
#pragma unroll
    for (int kf = 0; kf < 2; ++kf) {
      qa[kf].q[0] = *(const uint4*)(qrow + kf * 32 + 8 * g);
      qa[kf].q[1] = *(const uint4*)(qrow + kf * 32 + 16 + 8 * g);
    }
  }

  Acc8 oa[4];
  float mrow[8], lrow[8];
#pragma unroll
  for (int t = 0; t < 4; ++t)
#pragma unroll
    for (int r = 0; r < 8; ++r) oa[t].f[r] = 0.f;
#pragma unroll
  for (int r = 0; r < 8; ++r) { mrow[r] = -3.0e38f; lrow[r] = 0.f; }

  const int jmax = (q0 + 127) >> 6;    // last 64-key block needed by this WG
  for (int j = 0; j <= jmax; ++j) {
    const int kb = j * 64;
    // --- cooperative transposed V-block load into LDS ---
    {
      const int key = tid >> 2, hq = tid & 3;
      const __bf16* vrow = Vp + (size_t)(kb + key) * HD_ + hq * 16;
      Frag16 tmp;
      tmp.q[0] = *(const uint4*)(vrow);
      tmp.q[1] = *(const uint4*)(vrow + 8);
#pragma unroll
      for (int e = 0; e < 16; ++e) Vt[hq * 16 + e][key] = tmp.e[e];
    }
    __syncthreads();

    const bool compute = (kb <= qbase + 15);
    if (compute) {
      // scores init = rel_bias
      Acc8 sc[4];
#pragma unroll
      for (int t = 0; t < 4; ++t)
#pragma unroll
        for (int r = 0; r < 8; ++r)
          sc[t].f[r] = relp[(size_t)(qbase + r + 8 * g) * S_ + kb + t * 16 + ln];
      if (j < jmax)  // hint next rel_bias block toward the caches
        __builtin_prefetch(&relp[(size_t)(qbase + ln) * S_ + kb + 64], 0, 1);

      // S += Q * K^T   (K fragments contiguous per lane in [s,hd] layout)
#pragma unroll
      for (int t = 0; t < 4; ++t) {
        const __bf16* krow = Kp + (size_t)(kb + t * 16 + ln) * HD_;
#pragma unroll
        for (int kf = 0; kf < 2; ++kf) {
          Frag16 kbf;
          kbf.q[0] = *(const uint4*)(krow + kf * 32 + 8 * g);
          kbf.q[1] = *(const uint4*)(krow + kf * 32 + 16 + 8 * g);
          sc[t].v = wmma_bf16(qa[kf].v, kbf.v, sc[t].v);
        }
      }
      // causal + key-padding masks
#pragma unroll
      for (int t = 0; t < 4; ++t) {
        const int col = kb + t * 16 + ln;
        const bool padded = pad[b * S_ + col] != 0;
#pragma unroll
        for (int r = 0; r < 8; ++r) {
          const int row = qbase + r + 8 * g;
          if (col > row || padded) sc[t].f[r] = NEGINF;
        }
      }
      // online softmax (row = reg r + 8g; cols spread over 16-lane half-wave)
#pragma unroll
      for (int r = 0; r < 8; ++r) {
        float mx = fmaxf(fmaxf(sc[0].f[r], sc[1].f[r]),
                         fmaxf(sc[2].f[r], sc[3].f[r]));
#pragma unroll
        for (int msk = 1; msk < 16; msk <<= 1)
          mx = fmaxf(mx, __shfl_xor(mx, msk, 16));
        const float mnew  = fmaxf(mrow[r], mx);
        const float alpha = __expf(mrow[r] - mnew);
        float psum = 0.f;
#pragma unroll
        for (int t = 0; t < 4; ++t) {
          const float p = __expf(sc[t].f[r] - mnew);
          sc[t].f[r] = p; psum += p;
        }
#pragma unroll
        for (int msk = 1; msk < 16; msk <<= 1)
          psum += __shfl_xor(psum, msk, 16);
        mrow[r] = mnew;
        lrow[r] = lrow[r] * alpha + psum;
#pragma unroll
        for (int t = 0; t < 4; ++t) oa[t].f[r] *= alpha;
      }
      // P: D-layout -> A-layout round trip through per-wave LDS tile
#pragma unroll
      for (int t = 0; t < 4; ++t)
#pragma unroll
        for (int r = 0; r < 8; ++r)
          Pl[wv][r + 8 * g][t * 16 + ln] = (__bf16)sc[t].f[r];

      // O += P * V
#pragma unroll
      for (int kf = 0; kf < 2; ++kf) {
        Frag16 pa;
        const __bf16* prow = &Pl[wv][ln][0];
        pa.q[0] = *(const uint4*)(prow + kf * 32 + 8 * g);
        pa.q[1] = *(const uint4*)(prow + kf * 32 + 16 + 8 * g);
#pragma unroll
        for (int t = 0; t < 4; ++t) {
          Frag16 vb;
          const __bf16* vrow2 = &Vt[t * 16 + ln][0];
          vb.q[0] = *(const uint4*)(vrow2 + kf * 32 + 8 * g);
          vb.q[1] = *(const uint4*)(vrow2 + kf * 32 + 16 + 8 * g);
          oa[t].v = wmma_bf16(pa.v, vb.v, oa[t].v);
        }
      }
    }
    __syncthreads();   // Vt reused next iteration
  }

  // write O in [B,S,D] bf16
  __bf16* ows = (__bf16*)ows_u;
#pragma unroll
  for (int t = 0; t < 4; ++t) {
    const int d = h * HD_ + t * 16 + ln;
#pragma unroll
    for (int r = 0; r < 8; ++r) {
      const int s = qbase + r + 8 * g;
      const float v = oa[t].f[r] / lrow[r];
      ows[((size_t)b * S_ + s) * D_ + d] = (__bf16)v;
    }
  }
}

// ---------------------------------------------------------------------------
// Kernel 3: out-projection  O[8192,1024] @ Wout[1024,1024] + bout, zero padded.
// BK=64; A tile (already bf16) async-DMA'd into LDS.
// ---------------------------------------------------------------------------
__global__ __launch_bounds__(256) void out_gemm_kernel(
    const unsigned short* __restrict__ ows_u, const float* __restrict__ Wout,
    const float* __restrict__ bout, const unsigned char* __restrict__ pad,
    float* __restrict__ out) {
  __shared__ __align__(16) __bf16 As[128][80];
  __shared__ __align__(16) __bf16 Bt[128][80];

  const int tid  = threadIdx.x;
  const int lane = tid & 31, wv = tid >> 5;
  const int g = lane >> 4, ln = lane & 15;
  const int wr = wv >> 1, wc = wv & 1;
  const int n0 = blockIdx.x * 128;
  const int m0 = blockIdx.y * 128;
  const __bf16* O = (const __bf16*)ows_u;

  Acc8 acc[2][4];
#pragma unroll
  for (int i = 0; i < 2; ++i)
#pragma unroll
    for (int j = 0; j < 4; ++j)
#pragma unroll
      for (int r = 0; r < 8; ++r) acc[i][j].f[r] = 0.f;

  for (int k0 = 0; k0 < D_; k0 += 64) {
    {   // A tile: 128x64 bf16; 64B per thread
      const int row = tid >> 1, c = tid & 1;
      const __bf16* src = O + (size_t)(m0 + row) * D_ + k0 + c * 32;
      __bf16* dst = &As[row][c * 32];
#pragma unroll
      for (int qq = 0; qq < 4; ++qq) {
#if HAVE_ASYNC_LDS
        async_copy16(src + qq * 8, dst + qq * 8);
#else
        *(uint4*)(dst + qq * 8) = *(const uint4*)(src + qq * 8);
#endif
      }
    }
    {   // B tile: 64x128 fp32 -> bf16, transposed
      const int r0 = tid >> 5, c4 = tid & 31;
#pragma unroll
      for (int rr = 0; rr < 8; ++rr) {
        const int row = rr * 8 + r0;
        const float4 f = *(const float4*)(Wout + (size_t)(k0 + row) * D_ + n0 + c4 * 4);
        Bt[c4 * 4 + 0][row] = (__bf16)f.x;
        Bt[c4 * 4 + 1][row] = (__bf16)f.y;
        Bt[c4 * 4 + 2][row] = (__bf16)f.z;
        Bt[c4 * 4 + 3][row] = (__bf16)f.w;
      }
    }
#if HAVE_ASYNC_LDS
    wait_async();
#endif
    __syncthreads();

#pragma unroll
    for (int kk = 0; kk < 64; kk += 32) {
      Frag16 fa[2], fb[4];
#pragma unroll
      for (int ti = 0; ti < 2; ++ti) {
        const __bf16* base = &As[wr * 32 + ti * 16 + ln][kk];
        fa[ti].q[0] = *(const uint4*)(base + 8 * g);
        fa[ti].q[1] = *(const uint4*)(base + 16 + 8 * g);
      }
#pragma unroll
      for (int tj = 0; tj < 4; ++tj) {
        const __bf16* base = &Bt[wc * 64 + tj * 16 + ln][kk];
        fb[tj].q[0] = *(const uint4*)(base + 8 * g);
        fb[tj].q[1] = *(const uint4*)(base + 16 + 8 * g);
      }
#pragma unroll
      for (int ti = 0; ti < 2; ++ti)
#pragma unroll
        for (int tj = 0; tj < 4; ++tj)
          acc[ti][tj].v = wmma_bf16(fa[ti].v, fb[tj].v, acc[ti][tj].v);
    }
    __syncthreads();
  }

#pragma unroll
  for (int tj = 0; tj < 4; ++tj) {
    const int nc = n0 + wc * 64 + tj * 16 + ln;
    const float bias = bout[nc];
#pragma unroll
    for (int ti = 0; ti < 2; ++ti)
#pragma unroll
      for (int r = 0; r < 8; ++r) {
        const int mr = m0 + wr * 32 + ti * 16 + r + 8 * g;
        const int bb = mr >> 11, s = mr & 2047;
        float v = acc[ti][tj].f[r] + bias;
        if (pad[bb * S_ + s]) v = 0.f;
        out[(size_t)mr * D_ + nc] = v;
      }
  }
}

// ---------------------------------------------------------------------------
// Launch. Inputs (reference order): x, Wqkv, bqkv, Wout, bout, causal_mask,
// rel_bias, key_padding_mask. causal_mask is reproduced analytically.
// Workspace: Q,K,V,O bf16 buffers, 4 * 8388608 elems * 2B = 64 MB.
// ---------------------------------------------------------------------------
extern "C" void kernel_launch(void* const* d_in, const int* in_sizes, int n_in,
                              void* d_out, int out_size, void* d_ws, size_t ws_size,
                              hipStream_t stream) {
  const float* x    = (const float*)d_in[0];
  const float* Wqkv = (const float*)d_in[1];
  const float* bqkv = (const float*)d_in[2];
  const float* Wout = (const float*)d_in[3];
  const float* bout = (const float*)d_in[4];
  const float* rel  = (const float*)d_in[6];
  const unsigned char* pad = (const unsigned char*)d_in[7];
  float* out = (float*)d_out;

  const size_t elems = (size_t)B_ * H_ * S_ * HD_;   // 8388608 per tensor
  unsigned short* qws = (unsigned short*)d_ws;
  unsigned short* kws = qws + elems;
  unsigned short* vws = kws + elems;
  unsigned short* ows = vws + elems;

  qkv_gemm_kernel<<<dim3(N3D / 128, BS_ / 128), 256, 0, stream>>>(
      x, Wqkv, bqkv, qws, kws, vws);
  flash_attn_kernel<<<dim3(B_ * H_, S_ / 128), 256, 0, stream>>>(
      qws, kws, vws, rel, pad, ows);
  out_gemm_kernel<<<dim3(D_ / 128, BS_ / 128), 256, 0, stream>>>(
      ows, Wout, bout, pad, out);
}
